// MultiHeadAttention_38749194944967
// MI455X (gfx1250) — matrix operand order repaired and checked by reading, not compiled, not run on hardware
//
#include <hip/hip_runtime.h>
#include <hip/hip_bf16.h>
#include <math.h>

// Problem dims (match reference)
#define BB 2
#define SS 2048
#define HH 1024
#define NH 16
#define DD 64

typedef __attribute__((ext_vector_type(16))) _Float16 v16h;
typedef __attribute__((ext_vector_type(8)))  _Float16 v8h;
typedef __attribute__((ext_vector_type(8)))  float    v8f;

// ---------------------------------------------------------------------------
// WMMA fragment helpers (wave32, 16x16x32 f16 -> f32)
//
// A (16x32, row-major, stride lda):
//   lane = 16*half + m ; elements 0..7  = A[m][kbase + half*8 + j]
//                        elements 8..15 = A[m][kbase + 16 + half*8 + j]
// B (32x16, column-major storage Bcm[col][k], stride ldb):
//   lane = 16*half + n ; elements 0..7  = Bcm[nbase+n][kbase + half*8 + j]
//                        elements 8..15 = Bcm[nbase+n][kbase + 16 + half*8 + j]
// D (16x16 f32): element r -> row (r + 8*half), col (lane&15)
// ---------------------------------------------------------------------------

__device__ __forceinline__ v16h load_a16(const _Float16* __restrict__ Arow,
                                         size_t lda, int lane, int kbase) {
  int half = lane >> 4, m = lane & 15;
  const _Float16* p = Arow + (size_t)m * lda + kbase + half * 8;
  v8h lo = *(const v8h*)(p);
  v8h hi = *(const v8h*)(p + 16);
  v16h o;
#pragma unroll
  for (int i = 0; i < 8; ++i) { o[i] = lo[i]; o[8 + i] = hi[i]; }
  return o;
}

__device__ __forceinline__ v16h load_a16_from_f32(const float* __restrict__ Arow,
                                                  size_t lda, int lane, int kbase) {
  int half = lane >> 4, m = lane & 15;
  const float* p = Arow + (size_t)m * lda + kbase + half * 8;
  v16h o;
#pragma unroll
  for (int i = 0; i < 8; ++i) {
    o[i]     = (_Float16)p[i];
    o[8 + i] = (_Float16)p[16 + i];
  }
  return o;
}

__device__ __forceinline__ v16h load_b16(const _Float16* __restrict__ Bcm,
                                         size_t ldb, int lane, int nbase, int kbase) {
  int half = lane >> 4, n = lane & 15;
  const _Float16* p = Bcm + (size_t)(nbase + n) * ldb + kbase + half * 8;
  v8h lo = *(const v8h*)(p);
  v8h hi = *(const v8h*)(p + 16);
  v16h o;
#pragma unroll
  for (int i = 0; i < 8; ++i) { o[i] = lo[i]; o[8 + i] = hi[i]; }
  return o;
}

__device__ __forceinline__ void store_d_f32(float* __restrict__ Crow, size_t ldc,
                                            int lane, int nbase, v8f acc, float scale) {
  int half = lane >> 4, n = lane & 15;
#pragma unroll
  for (int r = 0; r < 8; ++r)
    Crow[(size_t)(r + 8 * half) * ldc + nbase + n] = acc[r] * scale;
}

__device__ __forceinline__ void store_d_f16(_Float16* __restrict__ Crow, size_t ldc,
                                            int lane, int nbase, v8f acc) {
  int half = lane >> 4, n = lane & 15;
#pragma unroll
  for (int r = 0; r < 8; ++r)
    Crow[(size_t)(r + 8 * half) * ldc + nbase + n] = (_Float16)acc[r];
}

// ---------------------------------------------------------------------------
// Prep kernels: f32->f16 convert, weight transposes (K-contiguous B layouts)
// ---------------------------------------------------------------------------

__global__ __launch_bounds__(256) void cvt_f16_kernel(const float* __restrict__ in,
                                                      _Float16* __restrict__ out,
                                                      size_t n) {
  for (size_t i = (size_t)blockIdx.x * blockDim.x + threadIdx.x; i < n;
       i += (size_t)gridDim.x * blockDim.x)
    out[i] = (_Float16)in[i];
}

// W [N,H,D] f32 -> Wt [N,D,H] f16
__global__ __launch_bounds__(256) void transpose_w_kernel(const float* __restrict__ W,
                                                          _Float16* __restrict__ Wt) {
  size_t total = (size_t)NH * HH * DD;
  for (size_t i = (size_t)blockIdx.x * blockDim.x + threadIdx.x; i < total;
       i += (size_t)gridDim.x * blockDim.x) {
    size_t n = i / ((size_t)HH * DD);
    size_t r = i % ((size_t)HH * DD);
    size_t h = r / DD, d = r % DD;
    Wt[n * (size_t)DD * HH + d * HH + h] = (_Float16)W[i];
  }
}

// Wo [H_in,H_out] f32 -> WoT [H_out,H_in] f16
__global__ __launch_bounds__(256) void transpose_wo_kernel(const float* __restrict__ Wo,
                                                           _Float16* __restrict__ WoT) {
  size_t total = (size_t)HH * HH;
  for (size_t idx = (size_t)blockIdx.x * blockDim.x + threadIdx.x; idx < total;
       idx += (size_t)gridDim.x * blockDim.x) {
    size_t hi = idx / HH, ho = idx % HH;
    WoT[ho * (size_t)HH + hi] = (_Float16)Wo[idx];
  }
}

// ---------------------------------------------------------------------------
// Projection GEMM: Out[b,n,s,d] = sum_h X[b,s,h] * Wt[n,d,h]
// One wave = one 16x64 strip (full D), 4 accumulators sharing each A fragment.
// vtrans=1 stores V transposed [B,N,D,S].
// Waves: B*N*(S/16) = 4096 -> 512 blocks of 8 waves.
// ---------------------------------------------------------------------------
__global__ __launch_bounds__(256) void proj_kernel(const _Float16* __restrict__ X,
                                                   const _Float16* __restrict__ Wt,
                                                   _Float16* __restrict__ Out,
                                                   int vtrans) {
  int w = blockIdx.x * 8 + (threadIdx.x >> 5);
  int lane = threadIdx.x & 31;
  int st = w & 127; w >>= 7;   // S tile (0..127)
  int n = w & 15;  int b = w >> 4;

  const _Float16* Arow = X + ((size_t)b * SS + (size_t)st * 16) * HH;
  const _Float16* Bcm  = Wt + (size_t)n * DD * HH;  // [D][H], ld=HH

  v8f acc[4] = {};
  for (int k = 0; k < HH; k += 32) {
    __builtin_prefetch(Arow + (size_t)(lane & 15) * HH + k + 64, 0, 0);
    v16h a = load_a16(Arow, HH, lane, k);
#pragma unroll
    for (int c = 0; c < 4; ++c) {
      v16h bf = load_b16(Bcm, HH, lane, c * 16, k);
      acc[c] = __builtin_amdgcn_wmma_f32_16x16x32_f16(false, a, false, bf,
                                                      (short)0, acc[c], false, false);
    }
  }

  if (!vtrans) {
    _Float16* Crow = Out + (((size_t)(b * NH + n)) * SS + (size_t)st * 16) * DD;
#pragma unroll
    for (int c = 0; c < 4; ++c) store_d_f16(Crow, DD, lane, c * 16, acc[c]);
  } else {
    _Float16* Cb = Out + ((size_t)(b * NH + n)) * DD * SS;  // [D][S]
    int half = lane >> 4, nc = lane & 15;
#pragma unroll
    for (int c = 0; c < 4; ++c)
#pragma unroll
      for (int r = 0; r < 8; ++r)
        Cb[(size_t)(c * 16 + nc) * SS + (size_t)st * 16 + r + 8 * half] =
            (_Float16)acc[c][r];
  }
}

// ---------------------------------------------------------------------------
// Scores: attn[b,n,i,j] = (1/8) * sum_d q[b,n,i,d] * k[b,n,j,d]   (f32 out)
// One wave = 16x64 strip of the score matrix (4 column tiles share A).
// Waves: B*N*(S/16)*(S/64) = 131072 -> 16384 blocks.
// ---------------------------------------------------------------------------
__global__ __launch_bounds__(256) void scores_kernel(const _Float16* __restrict__ q,
                                                     const _Float16* __restrict__ km,
                                                     float* __restrict__ attn) {
  int w = blockIdx.x * 8 + (threadIdx.x >> 5);
  int lane = threadIdx.x & 31;
  int ktg = w & 31; w >>= 5;   // 64-wide column group (0..31)
  int qt = w & 127; w >>= 7;   // 16-row tile (0..127)
  int n = w & 15;  int b = w >> 4;

  size_t head = (size_t)(b * NH + n);
  const _Float16* Arow = q + (head * SS + (size_t)qt * 16) * DD;
  const _Float16* Bcm  = km + head * SS * DD;  // col j -> row stride DD (contig d)

  v8f acc[4] = {};
#pragma unroll
  for (int k = 0; k < DD; k += 32) {
    v16h a = load_a16(Arow, DD, lane, k);
#pragma unroll
    for (int c = 0; c < 4; ++c) {
      v16h bf = load_b16(Bcm, DD, lane, ktg * 64 + c * 16, k);
      acc[c] = __builtin_amdgcn_wmma_f32_16x16x32_f16(false, a, false, bf,
                                                      (short)0, acc[c], false, false);
    }
  }
  float* Crow = attn + (head * SS + (size_t)qt * 16) * SS;
#pragma unroll
  for (int c = 0; c < 4; ++c)
    store_d_f32(Crow, SS, lane, ktg * 64 + c * 16, acc[c], 0.125f);  // 1/sqrt(64)
}

// ---------------------------------------------------------------------------
// Row softmax over attn [B*N*S rows, SS cols], in place
// ---------------------------------------------------------------------------
__global__ __launch_bounds__(256) void softmax_kernel(float* __restrict__ attn) {
  __shared__ float red[256];
  float* p = attn + (size_t)blockIdx.x * SS;
  int t = threadIdx.x;

  float m = -1e30f;
  for (int c = t; c < SS; c += 256) m = fmaxf(m, p[c]);
  red[t] = m; __syncthreads();
  for (int s = 128; s > 0; s >>= 1) {
    if (t < s) red[t] = fmaxf(red[t], red[t + s]);
    __syncthreads();
  }
  m = red[0]; __syncthreads();

  float sum = 0.f;
  for (int c = t; c < SS; c += 256) {
    float e = __expf(p[c] - m);
    p[c] = e;
    sum += e;
  }
  red[t] = sum; __syncthreads();
  for (int s = 128; s > 0; s >>= 1) {
    if (t < s) red[t] += red[t + s];
    __syncthreads();
  }
  float inv = 1.0f / red[0];
  for (int c = t; c < SS; c += 256) p[c] *= inv;
}

// ---------------------------------------------------------------------------
// ctx[b,n,s,d] = sum_j attn[b,n,s,j] * vT[b,n,d,j]   (f16 out)
// One wave = 16 rows x full D (4 accumulators share the converted A fragment).
// Waves: B*N*(S/16) = 4096 -> 512 blocks.
// ---------------------------------------------------------------------------
__global__ __launch_bounds__(256) void ctx_kernel(const float* __restrict__ attn,
                                                  const _Float16* __restrict__ vT,
                                                  _Float16* __restrict__ ctx) {
  int w = blockIdx.x * 8 + (threadIdx.x >> 5);
  int lane = threadIdx.x & 31;
  int st = w & 127; w >>= 7;
  int n = w & 15;  int b = w >> 4;

  size_t head = (size_t)(b * NH + n);
  const float* Arow = attn + (head * SS + (size_t)st * 16) * SS;
  const _Float16* Bcm = vT + head * (size_t)DD * SS;  // [D][S], ld=SS

  v8f acc[4] = {};
  for (int k = 0; k < SS; k += 32) {
    __builtin_prefetch(Arow + (size_t)(lane & 15) * SS + k + 64, 0, 0);
    v16h a = load_a16_from_f32(Arow, SS, lane, k);
#pragma unroll
    for (int c = 0; c < 4; ++c) {
      v16h bf = load_b16(Bcm, SS, lane, c * 16, k);
      acc[c] = __builtin_amdgcn_wmma_f32_16x16x32_f16(false, a, false, bf,
                                                      (short)0, acc[c], false, false);
    }
  }
  _Float16* Crow = ctx + (head * SS + (size_t)st * 16) * DD;
#pragma unroll
  for (int c = 0; c < 4; ++c) store_d_f16(Crow, DD, lane, c * 16, acc[c]);
}

// ---------------------------------------------------------------------------
// out[b,s,ho] = sum_{hi} concat[b,s,hi] * Wo[hi,ho]
// concat[b,s, n*64+d] = ctx[b,n,s,d] (gathered per head during A-frag load)
// One wave = 16 rows x 64 output columns (4 accumulators share A).
// Waves: (B*S/16)*(H/64) = 4096 -> 512 blocks.
// ---------------------------------------------------------------------------
__global__ __launch_bounds__(256) void outproj_kernel(const _Float16* __restrict__ ctx,
                                                      const _Float16* __restrict__ WoT,
                                                      float* __restrict__ out) {
  int w = blockIdx.x * 8 + (threadIdx.x >> 5);
  int lane = threadIdx.x & 31;
  int otg = w & 15;         // 64-wide output column group (0..15)
  int rt = w >> 4;          // row tile over B*S (0..255)

  int half = lane >> 4, rl = lane & 15;
  int row = rt * 16 + rl;
  int b = row / SS, s = row % SS;

  v8f acc[4] = {};
  for (int k = 0; k < HH; k += 32) {
    int k0 = k + half * 8;
    int k1 = k + 16 + half * 8;
    const _Float16* p0 =
        ctx + (((size_t)(b * NH + (k0 >> 6)) * SS + s) * DD + (k0 & 63));
    const _Float16* p1 =
        ctx + (((size_t)(b * NH + (k1 >> 6)) * SS + s) * DD + (k1 & 63));
    v8h lo = *(const v8h*)p0;
    v8h hi = *(const v8h*)p1;
    v16h a;
#pragma unroll
    for (int i = 0; i < 8; ++i) { a[i] = lo[i]; a[8 + i] = hi[i]; }
#pragma unroll
    for (int c = 0; c < 4; ++c) {
      v16h bf = load_b16(WoT, HH, lane, otg * 64 + c * 16, k);
      acc[c] = __builtin_amdgcn_wmma_f32_16x16x32_f16(false, a, false, bf,
                                                      (short)0, acc[c], false, false);
    }
  }
  float* Crow = out + (size_t)rt * 16 * HH;
#pragma unroll
  for (int c = 0; c < 4; ++c)
    store_d_f32(Crow, HH, lane, otg * 64 + c * 16, acc[c], 1.0f);
}

// ---------------------------------------------------------------------------
// Host-side launch
// ---------------------------------------------------------------------------
extern "C" void kernel_launch(void* const* d_in, const int* in_sizes, int n_in,
                              void* d_out, int out_size, void* d_ws, size_t ws_size,
                              hipStream_t stream) {
  const float* query = (const float*)d_in[0];
  const float* key_in = (const float*)d_in[1];
  const float* value = (const float*)d_in[2];
  const float* Wq = (const float*)d_in[3];
  const float* Wk = (const float*)d_in[4];
  const float* Wv = (const float*)d_in[5];
  const float* Wo = (const float*)d_in[6];

  float* out = (float*)d_out;                       // [B,S,H]
  float* attn = out + (size_t)BB * SS * HH;         // [B,N,S,S]

  char* ws = (char*)d_ws;
  const size_t MB = 1ull << 20;
  _Float16* Xq  = (_Float16*)(ws + 0 * MB);   // [B*S,H] f16   8 MB
  _Float16* Xk  = (_Float16*)(ws + 8 * MB);   //               8 MB
  _Float16* Xv  = (_Float16*)(ws + 16 * MB);  //               8 MB
  _Float16* Wqt = (_Float16*)(ws + 24 * MB);  // [N,D,H]       2 MB
  _Float16* Wkt = (_Float16*)(ws + 26 * MB);  //               2 MB
  _Float16* Wvt = (_Float16*)(ws + 28 * MB);  //               2 MB
  _Float16* WoT = (_Float16*)(ws + 30 * MB);  // [Ho,Hi]       2 MB
  _Float16* qh  = (_Float16*)(ws + 32 * MB);  // [B,N,S,D]     8 MB
  _Float16* kh  = (_Float16*)(ws + 40 * MB);  // [B,N,S,D]     8 MB
  _Float16* vT  = (_Float16*)(ws + 48 * MB);  // [B,N,D,S]     8 MB
  _Float16* ctx = (_Float16*)(ws + 56 * MB);  // [B,N,S,D]     8 MB

  dim3 blk(256);
  size_t nX = (size_t)BB * SS * HH;

  cvt_f16_kernel<<<1024, blk, 0, stream>>>(query, Xq, nX);
  cvt_f16_kernel<<<1024, blk, 0, stream>>>(key_in, Xk, nX);
  cvt_f16_kernel<<<1024, blk, 0, stream>>>(value, Xv, nX);
  transpose_w_kernel<<<512, blk, 0, stream>>>(Wq, Wqt);
  transpose_w_kernel<<<512, blk, 0, stream>>>(Wk, Wkt);
  transpose_w_kernel<<<512, blk, 0, stream>>>(Wv, Wvt);
  transpose_wo_kernel<<<512, blk, 0, stream>>>(Wo, WoT);

  // Projections: 4096 strip-waves each -> 512 blocks of 8 waves
  proj_kernel<<<512, blk, 0, stream>>>(Xq, Wqt, qh, 0);
  proj_kernel<<<512, blk, 0, stream>>>(Xk, Wkt, kh, 0);
  proj_kernel<<<512, blk, 0, stream>>>(Xv, Wvt, vT, 1);

  // Scores: 131072 strip-waves -> 16384 blocks
  scores_kernel<<<16384, blk, 0, stream>>>(qh, kh, attn);

  // Softmax: one block per row (B*N*S rows)
  softmax_kernel<<<BB * NH * SS, blk, 0, stream>>>(attn);

  // attn @ V : 4096 strip-waves -> 512 blocks
  ctx_kernel<<<512, blk, 0, stream>>>(attn, vT, ctx);

  // Output projection: 4096 strip-waves -> 512 blocks
  outproj_kernel<<<512, blk, 0, stream>>>(ctx, WoT, out);
}
